// FL_74088185856016
// MI455X (gfx1250) — compile-verified, matrix operand order checked by reading, not run on hardware
//
#include <hip/hip_runtime.h>
#include <hip/hip_bf16.h>

typedef __attribute__((ext_vector_type(16))) _Float16     v16h;
typedef __attribute__((ext_vector_type(8)))  float        v8f;
typedef __attribute__((ext_vector_type(4)))  unsigned int v4u;
typedef __attribute__((ext_vector_type(8)))  int          v8i;
typedef __attribute__((ext_vector_type(4)))  int          v4i;

#define EMB 128
#define KNEIGH 32
#define WELEMS (128 * 256)   // W_weight element count

// ---------------------------------------------------------------------------
// Kernel 0: one-shot convert W (f32, [128,256] row-major) -> f16 image in d_ws.
// Done once per launch instead of once per block; kernel 2 DMAs it via TDM.
// ---------------------------------------------------------------------------
__global__ __launch_bounds__(256) void convert_w_kernel(
    const float* __restrict__ W, _Float16* __restrict__ Wh)
{
    const int i = blockIdx.x * 256 + threadIdx.x;
    if (i < WELEMS) Wh[i] = (_Float16)W[i];
}

// ---------------------------------------------------------------------------
// Kernel 1: one wave32 per feature.
// lane l owns channels [4l, 4l+3]. alpha_k = <neigh_k, u> + mask, softmax over
// k=0..31 (register-resident, unrolled), then agg = sum_k alpha_k * neigh_k.
// Neighbor rows are read twice; the 51 MB table is L2-resident on MI455X.
// ---------------------------------------------------------------------------
__global__ __launch_bounds__(256) void attn_agg_kernel(
    const int*   __restrict__ adj,      // [F, 32]
    const float* __restrict__ emb_i,    // [N_NODE, 128]
    const float* __restrict__ u,        // [128]
    float*       __restrict__ agg,      // [F, 128] (workspace)
    int F)
{
    const int feat = (blockIdx.x * blockDim.x + threadIdx.x) >> 5;
    const int lane = threadIdx.x & 31;
    if (feat >= F) return;  // wave-uniform

    const float4 u4 = ((const float4*)u)[lane];
    const int idx_l = adj[(size_t)feat * KNEIGH + lane];  // one neighbor idx per lane

    float alpha[KNEIGH];
#pragma unroll
    for (int k = 0; k < KNEIGH; ++k) {
        const int idx = __shfl(idx_l, k, 32);
        const float4 r = ((const float4*)(emb_i + (size_t)idx * EMB))[lane];
        float p = r.x * u4.x + r.y * u4.y + r.z * u4.z + r.w * u4.w;
#pragma unroll
        for (int off = 16; off > 0; off >>= 1) p += __shfl_xor(p, off, 32);
        alpha[k] = p + (idx == 0 ? -10000.0f : 0.0f);
    }

    // softmax over k (every lane holds the full vector)
    float m = alpha[0];
#pragma unroll
    for (int k = 1; k < KNEIGH; ++k) m = fmaxf(m, alpha[k]);
    float s = 0.0f;
#pragma unroll
    for (int k = 0; k < KNEIGH; ++k) { alpha[k] = __expf(alpha[k] - m); s += alpha[k]; }
    const float inv = 1.0f / s;

    float4 acc = {0.0f, 0.0f, 0.0f, 0.0f};
#pragma unroll
    for (int k = 0; k < KNEIGH; ++k) {
        const int idx = __shfl(idx_l, k, 32);
        const float4 r = ((const float4*)(emb_i + (size_t)idx * EMB))[lane];
        const float w = alpha[k] * inv;
        acc.x += w * r.x; acc.y += w * r.y; acc.z += w * r.z; acc.w += w * r.w;
    }
    ((float4*)(agg + (size_t)feat * EMB))[lane] = acc;
}

// ---------------------------------------------------------------------------
// Kernel 2: gate = sigmoid([emb_f | agg] @ W^T + b); out = g*emb_f + (1-g)*agg
// GEMM [F,256]x[256,128] via v_wmma_f32_16x16x32_f16.
// Block = 256 threads = 8 waves; each wave computes 16 rows x 128 cols as
// 8 N-tiles, accumulating 8 K-chunks of 32 -> 64 WMMAs per wave.
// W (pre-converted to f16) is DMA'd into LDS by the Tensor Data Mover:
// one tensor_load_to_lds issued by wave 0, completion via s_wait_tensorcnt.
// A/B fragment element mapping (16-bit operands, wave32):
//   lane = {half: l>>4, idx: l&15}; element pair p (VGPR p):
//   K = (p<4 ? 2p : 2p+8) + (e&1) + 8*half   -- pairs are contiguous in K.
// ---------------------------------------------------------------------------
__global__ __launch_bounds__(256) void gate_gemm_kernel(
    const float*    __restrict__ embf,   // [F, 128]
    const float*    __restrict__ agg,    // [F, 128]
    const _Float16* __restrict__ Wh,     // [128, 256] row-major, f16
    const float*    __restrict__ bias,   // [128]
    float*          __restrict__ out,    // [F, 128]
    int F)
{
    __shared__ _Float16 Wl[WELEMS];  // 64 KB, B-operand source (B[c][n] = Wl[n*256+c])

    const int t    = threadIdx.x;
    const int wave = t >> 5;
    const int lane = t & 31;

    // ---- TDM: DMA the 64 KB f16 W image into LDS (1D tile, data_size=2B).
    if (wave == 0) {
        const unsigned long long ga = (unsigned long long)(uintptr_t)Wh;
        const unsigned lds = (unsigned)(uintptr_t)&Wl[0];
        v4u g0;
        g0.x = 1u;                                            // count=1, user desc
        g0.y = lds;                                           // lds_addr
        g0.z = (unsigned)(ga & 0xFFFFFFFFu);                  // global_addr[31:0]
        g0.w = (unsigned)((ga >> 32) & 0x01FFFFFFu)           // global_addr[56:32]
             | (2u << 30);                                    // type=2 (image)
        v8i g1;
        g1[0] = 0x00010000;        // wg_mask=0, data_size=1 (2 bytes), no pad/iter
        g1[1] = (int)0x80000000;   // tensor_dim0[15:0]=0x8000 -> dim0 = 32768
        g1[2] = 0x00010000;        // tensor_dim0[31:16]=0, tensor_dim1=1
        g1[3] = (int)0x80000000;   // tile_dim0 = 0x8000 (32768 elements)
        g1[4] = 0x00000001;        // tile_dim1 = 1, tile_dim2 = 0
        g1[5] = 32768;             // tensor_dim0_stride lo
        g1[6] = 0;                 // stride hi, tensor_dim1_stride lo
        g1[7] = 0;
        const v4i gz = {0, 0, 0, 0};
#if __clang_major__ >= 23
        const v8i gz8 = {0, 0, 0, 0, 0, 0, 0, 0};
        __builtin_amdgcn_tensor_load_to_lds(g0, g1, gz, gz, gz8, 0);
#else
        __builtin_amdgcn_tensor_load_to_lds(g0, g1, gz, gz, 0);
#endif
        __builtin_amdgcn_s_wait_tensorcnt(0);
    }
    __syncthreads();

    const int f0    = blockIdx.x * 128 + wave * 16;   // first row of this wave's strip
    const int midx  = lane & 15;                      // M (A) / N (B) index
    const int khalf = (lane >> 4) * 8;                // K offset for upper lane half
    int arow = f0 + midx; if (arow >= F) arow = F - 1;  // clamped A row

    v8f acc[8];
#pragma unroll
    for (int nt = 0; nt < 8; ++nt) acc[nt] = (v8f){0,0,0,0,0,0,0,0};

#pragma unroll
    for (int kc = 0; kc < 8; ++kc) {
        // ---- A fragment: cat row = [embf | agg]; this chunk lives in one array
        const float* asrc = (kc < 4)
            ? (embf + (size_t)arow * EMB + kc * 32)
            : (agg  + (size_t)arow * EMB + (kc - 4) * 32);
        v16h af;
#pragma unroll
        for (int p = 0; p < 8; ++p) {
            const int Kb = ((p < 4) ? 2 * p : 2 * p + 8) + khalf;   // even
            const float2 v = *(const float2*)(asrc + Kb);
            af[2 * p]     = (_Float16)v.x;
            af[2 * p + 1] = (_Float16)v.y;
        }
#pragma unroll
        for (int nt = 0; nt < 8; ++nt) {
            const _Float16* bsrc = &Wl[(nt * 16 + midx) * 256 + kc * 32];
            v16h bf;
#pragma unroll
            for (int p = 0; p < 8; ++p) {
                const int Kb = ((p < 4) ? 2 * p : 2 * p + 8) + khalf;
                bf[2 * p]     = bsrc[Kb];
                bf[2 * p + 1] = bsrc[Kb + 1];
            }
            acc[nt] = __builtin_amdgcn_wmma_f32_16x16x32_f16(
                false, af, false, bf, (short)0, acc[nt], false, false);
        }
    }

    // ---- Epilogue: bias + sigmoid gate + blend. C/D layout: VGPR r, lane l ->
    // M = r + 8*(l>=16), N = l&15.
    const int mbase = (lane >> 4) * 8;
#pragma unroll
    for (int nt = 0; nt < 8; ++nt) {
        const int col = nt * 16 + midx;
        const float b = bias[col];
#pragma unroll
        for (int r = 0; r < 8; ++r) {
            const int row = f0 + mbase + r;
            if (row < F) {
                const float z = acc[nt][r] + b;
                const float g = 1.0f / (1.0f + __expf(-z));
                const float e = embf[(size_t)row * EMB + col];
                const float a = agg [(size_t)row * EMB + col];
                out[(size_t)row * EMB + col] = g * e + (1.0f - g) * a;
            }
        }
    }
}

extern "C" void kernel_launch(void* const* d_in, const int* in_sizes, int n_in,
                              void* d_out, int out_size, void* d_ws, size_t ws_size,
                              hipStream_t stream) {
    const int*   adj   = (const int*)  d_in[0];   // [F, 32] int32
    const float* emb_i = (const float*)d_in[1];   // [N_NODE, 128]
    const float* embf  = (const float*)d_in[2];   // [F, 128]
    const float* u     = (const float*)d_in[3];   // [128, 1]
    const float* W     = (const float*)d_in[4];   // [128, 256]
    const float* bias  = (const float*)d_in[5];   // [128]
    float* out = (float*)d_out;

    const int F = in_sizes[0] / KNEIGH;

    // Workspace layout: agg [F*128 f32] | Wh [32768 f16]
    float*    agg = (float*)d_ws;
    _Float16* Wh  = (_Float16*)((char*)d_ws + (size_t)F * EMB * sizeof(float));

    // Stage 0: one-shot f16 conversion of W (TDM source image)
    convert_w_kernel<<<(WELEMS + 255) / 256, 256, 0, stream>>>(W, Wh);

    // Stage 1: attention + aggregation (1 wave per feature, 8 waves/block)
    const int blocks1 = (F + 7) / 8;
    attn_agg_kernel<<<blocks1, 256, 0, stream>>>(adj, emb_i, u, agg, F);

    // Stage 2: TDM-staged WMMA gate GEMM + blend (128 rows per block)
    const int blocks2 = (F + 127) / 128;
    gate_gemm_kernel<<<blocks2, 256, 0, stream>>>(embf, agg, Wh, bias, out, F);
}
